// SelfAttention_65481071394853
// MI455X (gfx1250) — compile-verified, matrix operand order checked by reading, not run on hardware
//
#include <hip/hip_runtime.h>
#include <hip/hip_bf16.h>

typedef __attribute__((ext_vector_type(16))) __bf16 bf16x16;
typedef __attribute__((ext_vector_type(8)))  __bf16 bf16x8;
typedef __attribute__((ext_vector_type(8)))  float  floatx8;

#define WAVES 8
#define LDT   264          // padded LDS row stride in halfs (256 + 8)

#define EP_BF16    0
#define EP_SIGMOID 1
#define EP_OUT     2

// ---------------------------------------------------------------------------
// WMMA operand loader (bf16 16x32 A-tile / 32x16 B-tile, ISA 7.12.2 layout):
// lane L<16 : row = row0+L, halfs {k..k+7} and {k+16..k+23}
// lane L>=16: row = row0+L-16, halfs {k+8..k+15} and {k+24..k+31}
// ---------------------------------------------------------------------------
__device__ __forceinline__ bf16x16 load_op(const __bf16* base, int ld, int k, int lane) {
    const __bf16* p = base + (lane & 15) * ld + k + ((lane >> 4) << 3);
    bf16x8 lo = *(const bf16x8*)p;
    bf16x8 hi = *(const bf16x8*)(p + 16);
    return __builtin_shufflevector(lo, hi, 0,1,2,3,4,5,6,7,8,9,10,11,12,13,14,15);
}

// ---------------------------------------------------------------------------
// C[m][n] = sum_k A[m][k] * B[n][k]   (K fixed = 256, both operands row-major
// along k -> contiguous 16B LDS/global loads per lane).
// Wave tile: 16 rows x 64 cols = 4 wmma accumulators.
// ---------------------------------------------------------------------------
__device__ __forceinline__ void gemm_ABt(
    const __bf16* __restrict__ A, int lda,
    const __bf16* __restrict__ B, int ldb,
    int M, int N,
    const float* __restrict__ rowBias,     // global f32 bias per output row (or null)
    __bf16*      __restrict__ dstLds, int ldd,
    float*       __restrict__ outG,        // EP_OUT: out + bc*65536 + g0
    const float* __restrict__ sgLds,       // EP_OUT: per-row bias in LDS
    int ep)
{
    const int lane = threadIdx.x & 31;
    const int wave = threadIdx.x >> 5;
    const int mt = M >> 4;
    const int nt = N >> 6;

    for (int t = wave; t < mt * nt; t += WAVES) {
        const int tm = (t % mt) << 4;
        const int tn = (t / mt) << 6;

        floatx8 acc[4];
        #pragma unroll
        for (int j = 0; j < 4; ++j) acc[j] = 0.0f;

        const __bf16* Arow = A + tm * lda;
        #pragma unroll
        for (int k = 0; k < 256; k += 32) {
            bf16x16 a = load_op(Arow, lda, k, lane);
            #pragma unroll
            for (int j = 0; j < 4; ++j) {
                bf16x16 b = load_op(B + (tn + j * 16) * ldb, ldb, k, lane);
                acc[j] = __builtin_amdgcn_wmma_f32_16x16x32_bf16(
                    false, a, false, b, (short)0, acc[j], false, false);
            }
        }

        // D layout: lane holds column (lane&15); VGPR r = row tm + r (+8 hi lanes)
        const int col   = lane & 15;
        const int rbase = (lane >> 4) << 3;

        #pragma unroll
        for (int j = 0; j < 4; ++j) {
            const int n0 = tn + j * 16 + col;
            if (ep == EP_OUT) {
                // out[p*256 + g], p = n0 (column), g = g0 + tm + rbase + r (contig in r)
                float* o = outG + n0 * 256 + tm + rbase;
                const int m0 = tm + rbase;
                float4 v0 = make_float4(acc[j][0] + sgLds[m0 + 0], acc[j][1] + sgLds[m0 + 1],
                                        acc[j][2] + sgLds[m0 + 2], acc[j][3] + sgLds[m0 + 3]);
                float4 v1 = make_float4(acc[j][4] + sgLds[m0 + 4], acc[j][5] + sgLds[m0 + 5],
                                        acc[j][6] + sgLds[m0 + 6], acc[j][7] + sgLds[m0 + 7]);
                *(float4*)(o)     = v0;
                *(float4*)(o + 4) = v1;
            } else if (ep == EP_SIGMOID) {
                #pragma unroll
                for (int r = 0; r < 8; ++r) {
                    float v = acc[j][r] * 0.0625f;            // / sqrt(HID)
                    v = 1.0f / (1.0f + __expf(-v));
                    dstLds[(tm + rbase + r) * ldd + n0] = (__bf16)v;
                }
            } else {
                #pragma unroll
                for (int r = 0; r < 8; ++r) {
                    float v = acc[j][r];
                    if (rowBias) v += rowBias[tm + rbase + r];
                    dstLds[(tm + rbase + r) * ldd + n0] = (__bf16)v;
                }
            }
        }
    }
}

// ---------------------------------------------------------------------------
// Kernel 0: convert weights f32 -> bf16 once: [Wq | Wk | Wv | Wv^T] in ws
// ---------------------------------------------------------------------------
__global__ void prep_weights(const float* __restrict__ Wq,
                             const float* __restrict__ Wk,
                             const float* __restrict__ Wv,
                             __bf16* __restrict__ wbf)
{
    const int i = blockIdx.x * 256 + threadIdx.x;      // 65536 elems / matrix
    wbf[i]          = (__bf16)Wq[i];
    wbf[i + 65536]  = (__bf16)Wk[i];
    wbf[i + 131072] = (__bf16)Wv[i];
    const int h = i >> 8, n = i & 255;
    wbf[196608 + n * 256 + h] = (__bf16)Wv[i];         // Wv^T
}

// ---------------------------------------------------------------------------
// Fused per-channel attention: one block per (b,c), 8 waves, ~305 KB LDS.
//   T   [256][LDT]  patches + pos-emb            (bf16)
//   KT  [256][LDT]  K^T = Wk * T^T + bk          (bf16)
//   per 32-col g-block: QT blk -> sigmoid(S^T) -> Z^T -> O^T (written to HBM)
// ---------------------------------------------------------------------------
__global__ void __launch_bounds__(256, 1) fused_attn(
    const float*  __restrict__ x,
    const __bf16* __restrict__ wbf,
    const float*  __restrict__ bq,
    const float*  __restrict__ bk,
    const float*  __restrict__ bv,
    float*        __restrict__ out)
{
    extern __shared__ __align__(16) char smemRaw[];
    __bf16* T   = (__bf16*)smemRaw;          // 256*LDT halfs
    __bf16* KT  = T  + 256 * LDT;            // 256*LDT
    __bf16* QZ  = KT + 256 * LDT;            // 32*LDT  (QT blk, later Zt blk)
    __bf16* At  = QZ + 32 * LDT;             // 32*LDT
    float*  sg  = (float*)(At + 32 * LDT);   // 32 floats
    float*  divtab = sg + 32;                // 128 floats: exp(-ln(1e4)/256 * 2j)

    const int tid = threadIdx.x;
    const int bc  = blockIdx.x;             // b*16 + c
    const int c   = bc & 15;

    const __bf16* Wq  = wbf;
    const __bf16* Wk  = wbf + 65536;
    const __bf16* WvT = wbf + 3 * 65536;

    // ---- phase 0: T[p][n] = x patch + positional embedding -----------------
    // Columns p (even) and p+1 share one angle: emb = {sin, cos}(pos*div[p>>1]),
    // and map to adjacent pixels -> pair-process with one __sincosf + float2.
    const float kDiv = -0.0359778921f;      // -ln(10000)/256
    if (tid < 128) divtab[tid] = __expf(kDiv * (float)(2 * tid));
    __syncthreads();

    const float* img = x + bc * 65536;
    for (int i2 = tid; i2 < 32768; i2 += 256) {
        const int i = i2 << 1;
        const int y = i >> 8, xc = i & 255;
        const int n = ((y >> 4) << 4) | (xc >> 4);     // patch index
        const int p = ((y & 15) << 4) | (xc & 15);     // in-patch (ed) index, even
        const float pos = (float)(c * 256 + n);
        const float ang = pos * divtab[p >> 1];
        float sn, cs;
        __sincosf(ang, &sn, &cs);
        const float2 v = *(const float2*)(img + i);
        T[p * LDT + n]       = (__bf16)(v.x + sn);
        T[(p + 1) * LDT + n] = (__bf16)(v.y + cs);
    }
    __syncthreads();

    // ---- phase 1: KT[h][p] = sum_n Wk[h][n] T[p][n] + bk[h] ---------------
    gemm_ABt(Wk, 256, T, LDT, 256, 256, bk, KT, LDT, nullptr, nullptr, EP_BF16);
    __syncthreads();

    float* outBase = out + bc * 65536;

    // ---- phase 2: stream over 32-column g-blocks ---------------------------
    for (int g0 = 0; g0 < 256; g0 += 32) {
        // QT[g][p] = Wq[g]·T^T + bq
        gemm_ABt(Wq + g0 * 256, 256, T, LDT, 32, 256, bq + g0,
                 QZ, LDT, nullptr, nullptr, EP_BF16);
        __syncthreads();

        // At[g][h] = sigmoid( (QT·KT^T)[g][h] / 16 )
        gemm_ABt(QZ, LDT, KT, LDT, 32, 256, nullptr,
                 At, LDT, nullptr, nullptr, EP_SIGMOID);
        __syncthreads();

        // sg[g] = sum_h bv[h]*At[g][h]   (wave 0; other waves start Zt)
        if (tid < 32) {
            float s = 0.0f;
            const __bf16* row = At + tid * LDT;
            for (int h = 0; h < 256; ++h) s += bv[h] * (float)row[h];
            sg[tid] = s;
        }

        // Zt[g][n] = At·WvT^T  (overwrites QZ; its readers synced above)
        gemm_ABt(At, LDT, WvT, 256, 32, 256, nullptr,
                 QZ, LDT, nullptr, nullptr, EP_BF16);
        __syncthreads();

        // Ot[g][p] = Zt·T^T + sg[g]  -> out[bc][p*256 + g0+g]  (f32, float4)
        gemm_ABt(QZ, LDT, T, LDT, 32, 256, nullptr,
                 nullptr, 0, outBase + g0, sg, EP_OUT);
        __syncthreads();
    }
}

extern "C" void kernel_launch(void* const* d_in, const int* in_sizes, int n_in,
                              void* d_out, int out_size, void* d_ws, size_t ws_size,
                              hipStream_t stream) {
    (void)in_sizes; (void)n_in; (void)out_size; (void)ws_size;
    const float* x  = (const float*)d_in[0];
    const float* Wq = (const float*)d_in[1];
    const float* bq = (const float*)d_in[2];
    const float* Wk = (const float*)d_in[3];
    const float* bk = (const float*)d_in[4];
    const float* Wv = (const float*)d_in[5];
    const float* bv = (const float*)d_in[6];

    __bf16* wbf = (__bf16*)d_ws;            // 4 * 65536 bf16 = 512 KB

    prep_weights<<<256, 256, 0, stream>>>(Wq, Wk, Wv, wbf);

    const size_t smem = (size_t)(2 * 256 * LDT + 2 * 32 * LDT) * sizeof(__bf16)
                      + (32 + 128) * sizeof(float);            // 304,768 B
    (void)hipFuncSetAttribute((const void*)fused_attn,
                              hipFuncAttributeMaxDynamicSharedMemorySize, (int)smem);
    fused_attn<<<512, 256, smem, stream>>>(x, wbf, bq, bk, bv, (float*)d_out);
}
